// GraphUpdater_43293270344180
// MI455X (gfx1250) — compile-verified, matrix operand order checked by reading, not run on hardware
//
#include <hip/hip_runtime.h>
#include <hip/hip_bf16.h>

typedef __attribute__((ext_vector_type(16))) _Float16 v16h;
typedef __attribute__((ext_vector_type(8)))  float    v8f;

// Problem constants
#define ZD 8
#define ND 512
#define KD 30
#define DD 128
#define ROWS_E (ZD*ND*KD)   // 122880
#define ROWS_N (ZD*ND)      // 4096
#define EPSN 1e-5f

// GEMM tiling: block = 128x128, 8 waves, wave tile = 32 rows x 64 cols
#define BM 128
#define BN 128
#define BK 32
#define LDA 40   // padded LDS stride in halves (conflict-free fragment reads)

// ---------------------------------------------------------------- loaders
// Loaders return two consecutive columns (col is always even; all sources are
// pairwise-contiguous at even columns).
struct PlainLoader {
  const float* A; int ldk;
  __device__ float2 load2(long row, int col) const {
    return *(const float2*)(A + row * (long)ldk + col);
  }
};

// conv1 implicit GEMM: row=(b,voxel), col=(ci,kd,kh,kw); latent (B,128,4,4,4)
// even col => kw = {0,1} => adjacent along W => contiguous pair.
struct Conv1Loader {
  const float* lat;
  __device__ float2 load2(long row, int col) const {
    long b = row >> 3; int vox = (int)(row & 7);
    int od = vox >> 2, oh = (vox >> 1) & 1, ow = vox & 1;
    int ci = col >> 3, kk = col & 7;
    int kd = kk >> 2, kh = (kk >> 1) & 1;          // kw = 0
    int d = 2*od + kd, h = 2*oh + kh, w = 2*ow;
    return *(const float2*)(lat + (b*128 + ci)*64 + d*16 + h*4 + w);
  }
};

// message-MLP layer1: row=(z,n,k) -> [V[z,n] | V[z,nbrs[z,n,k]] | edges[z,n,k]]
// 128-wide segments never split an even pair.
struct ConcatLoader {
  const float* V; const float* E; const long long* nbrs;
  __device__ float2 load2(long row, int col) const {
    long zn = row / KD;
    const float* p;
    if (col < DD)            p = V + zn*DD + col;
    else if (col < 2*DD) {
      long z = zn / ND;
      p = V + (z*ND + nbrs[row])*DD + (col - DD);
    } else                   p = E + row*DD + (col - 2*DD);
    return *(const float2*)p;
  }
};

// ------------------------------------------------------------- WMMA GEMM
// out[M,N] = act(A[M,K] @ W[K,N] + bias), W pre-transposed f16 as Wt[n*K+k].
// ACT: 0=none 1=silu 2=gelu(exact)
template<int ACT, typename AL>
__global__ __launch_bounds__(256) void gemm_wmma(
    AL al, const _Float16* __restrict__ Wt, const float* __restrict__ bias,
    float* __restrict__ out, int N, int K)
{
  __shared__ _Float16 As[BM * LDA];
  const int tid  = threadIdx.x;
  const int lane = tid & 31;
  const int wave = tid >> 5;        // 8 waves
  const int wr   = wave >> 1;       // wave row 0..3  (32 rows each)
  const int wc   = wave & 1;        // wave col 0..1  (64 cols each)
  const long rowBase = (long)blockIdx.x * BM;
  const int  colBase = blockIdx.y * BN + wc * 64;
  const int r = lane & 15, h = lane >> 4;

  v8f acc[2][4] = {};

  for (int k0 = 0; k0 < K; k0 += BK) {
    // stage A tile (f32 -> packed f16 pairs) into LDS
    for (int i = tid; i < BM*BK/2; i += 256) {
      int rr = i >> 4;              // 16 pairs per row
      int cc = (i & 15) << 1;
      float2 v = al.load2(rowBase + rr, k0 + cc);
      union { _Float16 hh[2]; uint32_t u; } pk;
      pk.hh[0] = (_Float16)v.x; pk.hh[1] = (_Float16)v.y;
      *(uint32_t*)(As + rr*LDA + cc) = pk.u;
    }
    __syncthreads();

    // B fragments straight from global (weights are L2-resident)
    v16h bfrag[4];
    #pragma unroll
    for (int ni = 0; ni < 4; ++ni) {
      int n = colBase + ni*16 + r;
      union { v16h v; uint32_t u[8]; } bu;
      const uint32_t* wp = (const uint32_t*)(Wt + (size_t)n * K + k0);
      #pragma unroll
      for (int e = 0; e < 8; ++e) {
        int kk = ((e & 3) << 1) + (h << 3) + ((e >> 2) << 4); // ISA 16-bit A/B K map
        bu.u[e] = wp[kk >> 1];
      }
      bfrag[ni] = bu.v;
    }

    #pragma unroll
    for (int mi = 0; mi < 2; ++mi) {
      union { v16h v; uint32_t u[8]; } au;
      const uint32_t* ap = (const uint32_t*)(As + (wr*32 + mi*16 + r) * LDA);
      #pragma unroll
      for (int e = 0; e < 8; ++e) {
        int kk = ((e & 3) << 1) + (h << 3) + ((e >> 2) << 4);
        au.u[e] = ap[kk >> 1];
      }
      #pragma unroll
      for (int ni = 0; ni < 4; ++ni)
        acc[mi][ni] = __builtin_amdgcn_wmma_f32_16x16x32_f16(
            false, au.v, false, bfrag[ni], (short)0, acc[mi][ni], false, false);
    }
    __syncthreads();
  }

  // epilogue: C lane map -> n = colBase+ni*16+r ; m = rowBase+wr*32+mi*16+8h+e
  #pragma unroll
  for (int mi = 0; mi < 2; ++mi)
  #pragma unroll
  for (int ni = 0; ni < 4; ++ni) {
    int n = colBase + ni*16 + r;
    float bv = bias ? bias[n] : 0.f;
    long m0 = rowBase + wr*32 + mi*16 + h*8;
    #pragma unroll
    for (int e = 0; e < 8; ++e) {
      float x = acc[mi][ni][e] + bv;
      if (ACT == 1)      x = x / (1.f + __expf(-x));
      else if (ACT == 2) x = 0.5f * x * (1.f + erff(x * 0.70710678118f));
      out[(m0 + e) * (long)N + n] = x;
    }
  }
}

// --------------------------------------------------------- weight prep
__global__ void cvt_f16(const float* __restrict__ w, _Float16* __restrict__ o, int n) {
  int i = blockIdx.x * 256 + threadIdx.x;
  if (i < n) o[i] = (_Float16)w[i];
}
__global__ void cvtT_f16(const float* __restrict__ w, _Float16* __restrict__ o, int Kd, int Nd) {
  int i = blockIdx.x * 256 + threadIdx.x;
  if (i < Kd * Nd) {
    int n = i / Kd, k = i - n * Kd;
    o[i] = (_Float16)w[(size_t)k * Nd + n];   // o[n*K+k] = W[k,n]
  }
}

// --------------------------------------------------------- norms & glue
// x: [b][vox][c] (GEMM output). GN over (16ch x 8vox)=128 per group, silu,
// write transposed [b][c][vox] so conv2's implicit-GEMM A is contiguous.
__global__ void gn1_silu(const float* __restrict__ x, const float* __restrict__ g,
                         const float* __restrict__ b, float* __restrict__ o) {
  __shared__ float ssum[128], ssq[128], gm[8], gv[8];
  int c = threadIdx.x;
  const float* xb = x + (size_t)blockIdx.x * 1024;
  float v[8], s = 0.f, q = 0.f;
  #pragma unroll
  for (int vox = 0; vox < 8; ++vox) { float t = xb[vox*128 + c]; v[vox]=t; s+=t; q+=t*t; }
  ssum[c] = s; ssq[c] = q; __syncthreads();
  int gi = c >> 4;
  if ((c & 15) == 0) {
    float S=0.f, Q=0.f;
    for (int i = 0; i < 16; ++i) { S += ssum[gi*16+i]; Q += ssq[gi*16+i]; }
    float m = S * (1.f/128.f);
    gm[gi] = m; gv[gi] = Q * (1.f/128.f) - m*m;
  }
  __syncthreads();
  float m = gm[gi], inv = rsqrtf(gv[gi] + EPSN);
  float gg = g[c], bb = b[c];
  float* ob = o + (size_t)blockIdx.x * 1024;
  #pragma unroll
  for (int vox = 0; vox < 8; ++vox) {
    float y = (v[vox] - m) * inv * gg + bb;
    ob[c*8 + vox] = y / (1.f + __expf(-y));
  }
}

__global__ void gn2_silu(const float* __restrict__ x, const float* __restrict__ g,
                         const float* __restrict__ b, float* __restrict__ o) {
  __shared__ float ssum[128], ssq[128], gm[8], gv[8];
  int c = threadIdx.x;
  size_t base = (size_t)blockIdx.x * 128;
  float t = x[base + c];
  ssum[c] = t; ssq[c] = t*t; __syncthreads();
  int gi = c >> 4;
  if ((c & 15) == 0) {
    float S=0.f, Q=0.f;
    for (int i = 0; i < 16; ++i) { S += ssum[gi*16+i]; Q += ssq[gi*16+i]; }
    float m = S * (1.f/16.f);
    gm[gi] = m; gv[gi] = Q * (1.f/16.f) - m*m;
  }
  __syncthreads();
  float y = (t - gm[gi]) * rsqrtf(gv[gi] + EPSN) * g[c] + b[c];
  o[base + c] = y / (1.f + __expf(-y));
}

__global__ void ew_add(float* __restrict__ a, const float* __restrict__ b, int n) {
  int i = blockIdx.x * 256 + threadIdx.x;
  if (i < n) a[i] += b[i];
}

__global__ void film_V(const float* __restrict__ gb, const float* __restrict__ nodes,
                       float* __restrict__ V) {
  int row = blockIdx.x, c = threadIdx.x;
  V[(size_t)row*DD + c] = gb[(size_t)row*2*DD + c] * nodes[(size_t)row*DD + c]
                        + gb[(size_t)row*2*DD + DD + c];
}

__device__ __forceinline__ void blk_ln_write(float s, int c, const float* g,
                                             const float* b, float* dst) {
  __shared__ float ssum[128], ssq[128], sm, sv;
  ssum[c] = s; ssq[c] = s*s; __syncthreads();
  if (c == 0) {
    float S=0.f, Q=0.f;
    for (int i = 0; i < 128; ++i) { S += ssum[i]; Q += ssq[i]; }
    float m = S * (1.f/128.f);
    sm = m; sv = Q * (1.f/128.f) - m*m;
  }
  __syncthreads();
  dst[c] = (s - sm) * rsqrtf(sv + EPSN) * g[c] + b[c];
}

// V2 = LN(Vin + sum_k Mv*mask)
__global__ void msg_reduce_ln(const float* __restrict__ Mv, const float* __restrict__ mask,
                              const float* __restrict__ Vin, const float* __restrict__ g,
                              const float* __restrict__ b, float* __restrict__ Vout) {
  int row = blockIdx.x, c = threadIdx.x;
  float s = Vin[(size_t)row*DD + c];
  const float* mr = Mv + (size_t)row * KD * DD;
  const float* mk = mask + (size_t)row * KD;
  for (int k = 0; k < KD; ++k) s += mr[k*DD + c] * mk[k];
  blk_ln_write(s, c, g, b, Vout + (size_t)row*DD);
}

// out = LN(res + x)
__global__ void row_ln(const float* __restrict__ x, const float* __restrict__ res,
                       const float* __restrict__ g, const float* __restrict__ b,
                       float* __restrict__ o) {
  int row = blockIdx.x, c = threadIdx.x;
  float s = x[(size_t)row*DD + c] + res[(size_t)row*DD + c];
  blk_ln_write(s, c, g, b, o + (size_t)row*DD);
}

// E = LN(edges + Me*mask)
__global__ void edge_ln(const float* __restrict__ Me, const float* __restrict__ edges,
                        const float* __restrict__ mask, const float* __restrict__ g,
                        const float* __restrict__ b, float* __restrict__ E) {
  int row = blockIdx.x, c = threadIdx.x;
  float s = edges[(size_t)row*DD + c] + Me[(size_t)row*DD + c] * mask[row];
  blk_ln_write(s, c, g, b, E + (size_t)row*DD);
}

// ---------------------------------------------------------------- host
extern "C" void kernel_launch(void* const* d_in, const int* in_sizes, int n_in,
                              void* d_out, int out_size, void* d_ws, size_t ws_size,
                              hipStream_t stream) {
  (void)in_sizes; (void)n_in; (void)out_size;
  const float*      latent = (const float*)d_in[0];
  const float*      t_in   = (const float*)d_in[1];
  const float*      nodes  = (const float*)d_in[2];
  const float*      edges  = (const float*)d_in[3];
  const long long*  nbrs   = (const long long*)d_in[4];   // jnp.int64
  const float*      nmask  = (const float*)d_in[5];
  // params (dict insertion order, lists expanded)
  const float* cw1   = (const float*)d_in[6];
  const float* gn1g  = (const float*)d_in[7];
  const float* gn1b  = (const float*)d_in[8];
  const float* cw2   = (const float*)d_in[9];
  const float* gn2g  = (const float*)d_in[10];
  const float* gn2b  = (const float*)d_in[11];
  const float* cw3   = (const float*)d_in[12];
  const float* cb3   = (const float*)d_in[13];
  const float* fW0   = (const float*)d_in[14];
  const float* fW1   = (const float*)d_in[15];
  const float* fW2   = (const float*)d_in[16];
  const float* fb0   = (const float*)d_in[17];
  const float* fb1   = (const float*)d_in[18];
  const float* fb2   = (const float*)d_in[19];
  const float* nW0   = (const float*)d_in[20];
  const float* nW1   = (const float*)d_in[21];
  const float* nW2   = (const float*)d_in[22];
  const float* nb0   = (const float*)d_in[23];
  const float* nb1   = (const float*)d_in[24];
  const float* nb2   = (const float*)d_in[25];
  const float* lnnmg = (const float*)d_in[26];
  const float* lnnmb = (const float*)d_in[27];
  const float* xW0   = (const float*)d_in[28];
  const float* xW1   = (const float*)d_in[29];
  const float* xb0   = (const float*)d_in[30];
  const float* xb1   = (const float*)d_in[31];
  const float* lnffg = (const float*)d_in[32];
  const float* lnffb = (const float*)d_in[33];
  const float* eW0   = (const float*)d_in[34];
  const float* eW1   = (const float*)d_in[35];
  const float* eW2   = (const float*)d_in[36];
  const float* eb0   = (const float*)d_in[37];
  const float* eb1   = (const float*)d_in[38];
  const float* eb2   = (const float*)d_in[39];
  const float* lnemg = (const float*)d_in[40];
  const float* lnemb = (const float*)d_in[41];

  // workspace layout
  char* W = (char*)d_ws;
  size_t off = 0;
  auto alloc = [&](size_t bytes) -> char* {
    char* p = W + off;
    off = (off + bytes + 255) & ~(size_t)255;
    return p;
  };
  _Float16* cw1h = (_Float16*)alloc(131072*2);
  _Float16* cw2h = (_Float16*)alloc(131072*2);
  _Float16* cw3h = (_Float16*)alloc(16384*2);
  _Float16* fW0t = (_Float16*)alloc(16384*2);
  _Float16* fW1t = (_Float16*)alloc(16384*2);
  _Float16* fW2t = (_Float16*)alloc(32768*2);
  _Float16* nW0t = (_Float16*)alloc(49152*2);
  _Float16* nW1t = (_Float16*)alloc(16384*2);
  _Float16* nW2t = (_Float16*)alloc(16384*2);
  _Float16* xW0t = (_Float16*)alloc(65536*2);
  _Float16* xW1t = (_Float16*)alloc(65536*2);
  _Float16* eW0t = (_Float16*)alloc(49152*2);
  _Float16* eW1t = (_Float16*)alloc(16384*2);
  _Float16* eW2t = (_Float16*)alloc(16384*2);
  float* x1  = (float*)alloc((size_t)32768*128*4);
  float* x1t = (float*)alloc((size_t)32768*128*4);
  float* x2  = (float*)alloc((size_t)ROWS_N*128*4);
  float* x2n = (float*)alloc((size_t)ROWS_N*128*4);
  float* lat = (float*)alloc((size_t)ROWS_N*128*4);
  float* l1  = (float*)alloc((size_t)ROWS_N*128*4);
  float* l2  = (float*)alloc((size_t)ROWS_N*128*4);
  float* gb  = (float*)alloc((size_t)ROWS_N*256*4);
  float* Vf  = (float*)alloc((size_t)ROWS_N*128*4);
  float* V2  = (float*)alloc((size_t)ROWS_N*128*4);
  float* f1  = (float*)alloc((size_t)ROWS_N*512*4);
  float* f2  = (float*)alloc((size_t)ROWS_N*128*4);
  float* h1  = (float*)alloc((size_t)ROWS_E*128*4);
  float* h2  = (float*)alloc((size_t)ROWS_E*128*4);
  if (off > ws_size) return;  // workspace too small; bail deterministically

  float* Vout = (float*)d_out;            // Z*N*D floats
  float* Eout = (float*)d_out + (size_t)ROWS_N*DD;

  // ---- weight prep (f16 / transposed f16)
  auto cd = [&](const float* src, _Float16* dst, int n) {
    cvt_f16<<<(n+255)/256, 256, 0, stream>>>(src, dst, n);
  };
  auto ct = [&](const float* src, _Float16* dst, int Kd, int Nd) {
    cvtT_f16<<<(Kd*Nd+255)/256, 256, 0, stream>>>(src, dst, Kd, Nd);
  };
  cd(cw1, cw1h, 131072); cd(cw2, cw2h, 131072); cd(cw3, cw3h, 16384);
  ct(fW0, fW0t, 128,128); ct(fW1, fW1t, 128,128); ct(fW2, fW2t, 128,256);
  ct(nW0, nW0t, 384,128); ct(nW1, nW1t, 128,128); ct(nW2, nW2t, 128,128);
  ct(xW0, xW0t, 128,512); ct(xW1, xW1t, 512,128);
  ct(eW0, eW0t, 384,128); ct(eW1, eW1t, 128,128); ct(eW2, eW2t, 128,128);

  // ---- encoder convs
  gemm_wmma<0, Conv1Loader><<<dim3(32768/BM, 1), 256, 0, stream>>>(
      Conv1Loader{latent}, cw1h, nullptr, x1, 128, 1024);
  gn1_silu<<<ROWS_N, 128, 0, stream>>>(x1, gn1g, gn1b, x1t);
  gemm_wmma<0, PlainLoader><<<dim3(ROWS_N/BM, 1), 256, 0, stream>>>(
      PlainLoader{x1t, 1024}, cw2h, nullptr, x2, 128, 1024);
  gn2_silu<<<ROWS_N, 128, 0, stream>>>(x2, gn2g, gn2b, x2n);
  gemm_wmma<0, PlainLoader><<<dim3(ROWS_N/BM, 1), 256, 0, stream>>>(
      PlainLoader{x2n, 128}, cw3h, cb3, lat, 128, 128);
  ew_add<<<(ROWS_N*128+255)/256, 256, 0, stream>>>(lat, t_in, ROWS_N*128);

  // ---- FiLM
  gemm_wmma<1, PlainLoader><<<dim3(ROWS_N/BM, 1), 256, 0, stream>>>(
      PlainLoader{lat, 128}, fW0t, fb0, l1, 128, 128);
  gemm_wmma<1, PlainLoader><<<dim3(ROWS_N/BM, 1), 256, 0, stream>>>(
      PlainLoader{l1, 128}, fW1t, fb1, l2, 128, 128);
  gemm_wmma<0, PlainLoader><<<dim3(ROWS_N/BM, 2), 256, 0, stream>>>(
      PlainLoader{l2, 128}, fW2t, fb2, gb, 256, 128);
  film_V<<<ROWS_N, 128, 0, stream>>>(gb, nodes, Vf);

  // ---- message MLP + aggregate + LN
  gemm_wmma<2, ConcatLoader><<<dim3(ROWS_E/BM, 1), 256, 0, stream>>>(
      ConcatLoader{Vf, edges, nbrs}, nW0t, nb0, h1, 128, 384);
  gemm_wmma<2, PlainLoader><<<dim3(ROWS_E/BM, 1), 256, 0, stream>>>(
      PlainLoader{h1, 128}, nW1t, nb1, h2, 128, 128);
  gemm_wmma<0, PlainLoader><<<dim3(ROWS_E/BM, 1), 256, 0, stream>>>(
      PlainLoader{h2, 128}, nW2t, nb2, h1, 128, 128);
  msg_reduce_ln<<<ROWS_N, 128, 0, stream>>>(h1, nmask, Vf, lnnmg, lnnmb, V2);

  // ---- FFN + LN -> V output
  gemm_wmma<2, PlainLoader><<<dim3(ROWS_N/BM, 4), 256, 0, stream>>>(
      PlainLoader{V2, 128}, xW0t, xb0, f1, 512, 128);
  gemm_wmma<0, PlainLoader><<<dim3(ROWS_N/BM, 1), 256, 0, stream>>>(
      PlainLoader{f1, 512}, xW1t, xb1, f2, 128, 512);
  row_ln<<<ROWS_N, 128, 0, stream>>>(f2, V2, lnffg, lnffb, Vout);

  // ---- edge MLP + LN -> E output (gathers from final V in d_out)
  gemm_wmma<2, ConcatLoader><<<dim3(ROWS_E/BM, 1), 256, 0, stream>>>(
      ConcatLoader{Vout, edges, nbrs}, eW0t, eb0, h1, 128, 384);
  gemm_wmma<2, PlainLoader><<<dim3(ROWS_E/BM, 1), 256, 0, stream>>>(
      PlainLoader{h1, 128}, eW1t, eb1, h2, 128, 128);
  gemm_wmma<0, PlainLoader><<<dim3(ROWS_E/BM, 1), 256, 0, stream>>>(
      PlainLoader{h2, 128}, eW2t, eb2, h1, 128, 128);
  edge_ln<<<ROWS_E, 128, 0, stream>>>(h1, edges, nmask, lnemg, lnemb, Eout);
}
